// FeatureRecoveryAdvanced_23553600651342
// MI455X (gfx1250) — matrix operand order repaired and checked by reading, not compiled, not run on hardware
//
#include <hip/hip_runtime.h>

// ---------------------------------------------------------------------------
// FeatureRecoveryAdvanced for MI455X (gfx1250, wave32, WMMA)
//   out = relu(quant(quant(x)@Wq) @ Wq^T + (x@Wc) @ Wc^T + b)
// GEMMs: v_wmma_f32_16x16x32_f16, double-buffered LDS fed by
// global_load_async_to_lds_b128 (ASYNCcnt pipeline), 128x128 block tile.
// Workspace usage ~321 MB.
// ---------------------------------------------------------------------------

typedef _Float16 half_t;
typedef __attribute__((ext_vector_type(16))) _Float16 v16h;
typedef __attribute__((ext_vector_type(8)))  float    v8f;

constexpr int Nn = 1024;        // N  (rows of W, output cols)
constexpr int Mm = 8192;        // M  (cols of W)
constexpr int Bb = 8192;        // batch
constexpr int Mh = Mm / 2;      // 4096 (expanded / continuous halves)

// ---------------------------------------------------------------- utilities
__device__ inline float waveMaxF(float v) {
  #pragma unroll
  for (int off = 16; off > 0; off >>= 1)
    v = fmaxf(v, __shfl_xor(v, off, 32));
  return v;
}

// CDNA5 async DMA: 16 bytes global -> LDS, tracked by ASYNCcnt
__device__ __forceinline__ void async_g2l_b128(const half_t* g, half_t* l) {
  unsigned loff = (unsigned)(unsigned long long)
      (__attribute__((address_space(3))) half_t*)l;
  asm volatile("global_load_async_to_lds_b128 %0, %1, off"
               :: "v"(loff), "v"((const void*)g) : "memory");
}

__global__ void k_init_scalars(float* s) { s[0] = 0.f; s[1] = 0.f; }

// per-tensor absmax -> atomicMax on IEEE bits (all values >= 0, bit-monotone)
__global__ void k_absmax(const float* __restrict__ x, long n, unsigned* out) {
  long stride = (long)gridDim.x * blockDim.x;
  float m = 0.f;
  for (long i = blockIdx.x * (long)blockDim.x + threadIdx.x; i < n; i += stride)
    m = fmaxf(m, fabsf(x[i]));
  m = waveMaxF(m);
  __shared__ float sm[8];
  if ((threadIdx.x & 31) == 0) sm[threadIdx.x >> 5] = m;
  __syncthreads();
  if (threadIdx.x == 0) {
    float b = sm[0];
    for (int j = 1; j < (int)(blockDim.x >> 5); ++j) b = fmaxf(b, sm[j]);
    atomicMax(out, __float_as_uint(b));
  }
}

// activation fake-quant: q = clip(round(x*s),-128,127)/s  (s = 127/absmax)
__global__ void k_act_quant(const float* __restrict__ x,
                            const float* __restrict__ amax,
                            half_t* __restrict__ xq,
                            half_t* __restrict__ xplain, long n) {
  float a = fmaxf(amax[0], 1e-8f);
  float scale = 127.f / a;
  float inv   = a / 127.f;
  long stride = (long)gridDim.x * blockDim.x;
  for (long i = blockIdx.x * (long)blockDim.x + threadIdx.x; i < n; i += stride) {
    float v = x[i];
    float q = rintf(v * scale);
    q = fminf(fmaxf(q, -128.f), 127.f);
    xq[i] = (half_t)(q * inv);
    if (xplain) xplain[i] = (half_t)v;
  }
}

// ------------------------------------------------------------ weight quant
// one wave32 per 32-element block of W_exp = W[:, idx]  (row-major [Nn][Mh])
// stores both orientations: Wq [Nn][Mh] and WqT [Mh][Nn]
__global__ void k_weight_quant(const float* __restrict__ W,
                               const int* __restrict__ idx,
                               half_t* __restrict__ Wq,
                               half_t* __restrict__ WqT) {
  int wid  = (int)((blockIdx.x * blockDim.x + threadIdx.x) >> 5);
  int lane = threadIdx.x & 31;
  long e   = (long)wid * 32 + lane;
  int row  = (int)(e / Mh);
  int col  = (int)(e % Mh);
  float w  = W[(long)row * Mm + idx[col]];

  float s  = fmaxf(waveMaxF(fabsf(w)), 1e-8f);
  float nb = w / s;
  float anb = fabsf(nb);

  // top-1 (value, index), tie -> lower index (matches jax top_k)
  float v1 = anb; int i1 = lane;
  #pragma unroll
  for (int off = 16; off > 0; off >>= 1) {
    float ov = __shfl_xor(v1, off, 32);
    int   oi = __shfl_xor(i1, off, 32);
    if (ov > v1 || (ov == v1 && oi < i1)) { v1 = ov; i1 = oi; }
  }
  // top-2: exclude i1
  float v2 = (lane == i1) ? -1.f : anb;
  int   i2 = (lane == i1) ? 1024 : lane;
  #pragma unroll
  for (int off = 16; off > 0; off >>= 1) {
    float ov = __shfl_xor(v2, off, 32);
    int   oi = __shfl_xor(i2, off, 32);
    if (ov > v2 || (ov == v2 && oi < i2)) { v2 = ov; i2 = oi; }
  }
  bool top = (lane == i1) || (lane == i2);

  float q4 = rintf(nb * 7.f) * (1.f / 7.f);
  float sgn = (nb > 0.f) ? 1.f : ((nb < 0.f) ? -1.f : 0.f);
  float q2 = sgn * (anb > 0.66f ? 1.f : (1.f / 3.f));
  half_t qv = (half_t)((top ? q4 : q2) * s);
  Wq[e] = qv;
  WqT[(long)col * Nn + row] = qv;
}

// gather continuous half of W to f16 in both orientations
__global__ void k_weight_cont(const float* __restrict__ W,
                              const int* __restrict__ idx_comp,
                              half_t* __restrict__ Wc,
                              half_t* __restrict__ WcT) {
  long n = (long)Nn * Mh;
  long stride = (long)gridDim.x * blockDim.x;
  for (long i = blockIdx.x * (long)blockDim.x + threadIdx.x; i < n; i += stride) {
    int row = (int)(i / Mh), c = (int)(i % Mh);
    half_t v = (half_t)W[(long)row * Mm + idx_comp[c]];
    Wc[i] = v;
    WcT[(long)c * Nn + row] = v;
  }
}

// ------------------------------------------------------------------- GEMM
// C[M x Ncols] = sum_p  A_p[M x K] * B_p^T,  f16 operands, f32 accum.
// B_p given transposed: element (k,n) = Bt_p[n*K + k]  (k-contiguous rows)
// mode 0: store f32 C     mode 1: store f16 C     mode 2: relu(C+bias) f32
// Block tile 128x128, 256 threads (8 wave32), wave grid 4(M) x 2(N),
// per-wave 2x4 fragments; double-buffered LDS filled by async b128 DMA.
__launch_bounds__(256)
__global__ void k_gemm_wmma(const half_t* __restrict__ A1,
                            const half_t* __restrict__ B1,
                            const half_t* __restrict__ A2,
                            const half_t* __restrict__ B2,
                            int K,
                            float* __restrict__ Cf, half_t* __restrict__ Ch,
                            const float* __restrict__ bias,
                            float* __restrict__ Out, int ldc, int mode) {
  // [buf][row/n][k] ; stride 40 halves = 80 B (16B-aligned, bank-spread)
  __shared__ __attribute__((aligned(16))) half_t As[2][128][40];
  __shared__ __attribute__((aligned(16))) half_t Bs[2][128][40];

  const int tid  = threadIdx.x;
  const int lane = tid & 31;
  const int wave = tid >> 5;
  const int wm   = wave >> 1;        // 0..3 : wave row group (32 rows)
  const int wn   = wave & 1;         // 0..1 : wave col group (64 cols)
  const int kgrp = lane >> 4;        // 0/1
  const int l16  = lane & 15;

  const long rowBlock = (long)blockIdx.y * 128;
  const int  colBlock = blockIdx.x * 128;

  const int tilesK = K >> 5;                       // K/32
  const int T = (A2 != nullptr) ? 2 * tilesK : tilesK;

  // issue one stage: 4 async b128 per thread (2 A units + 2 B units)
  auto issue = [&](int t, int buf) {
    const half_t* A  = (t >= tilesK) ? A2 : A1;
    const half_t* Bg = (t >= tilesK) ? B2 : B1;
    const int k0 = (t >= tilesK ? t - tilesK : t) << 5;
    #pragma unroll
    for (int h = 0; h < 2; ++h) {
      const int u   = tid + h * 256;               // 0..511
      const int r   = u >> 2;                      // 0..127
      const int seg = u & 3;                       // 16B segment in the row
      async_g2l_b128(A  + (rowBlock + r) * (long)K + k0 + seg * 8,
                     &As[buf][r][seg * 8]);
      async_g2l_b128(Bg + (long)(colBlock + r) * K + k0 + seg * 8,
                     &Bs[buf][r][seg * 8]);
    }
  };

  v8f acc[2][4];
  #pragma unroll
  for (int mi = 0; mi < 2; ++mi)
    #pragma unroll
    for (int ni = 0; ni < 4; ++ni) { v8f z = {}; acc[mi][ni] = z; }

  issue(0, 0);
  for (int t = 0; t < T; ++t) {
    const int buf = t & 1;
    if (t + 1 < T) {
      issue(t + 1, buf ^ 1);
      // stage t's 4 DMAs done (in-order), stage t+1's may stay in flight
      asm volatile("s_wait_asynccnt 0x4" ::: "memory");
    } else {
      asm volatile("s_wait_asynccnt 0x0" ::: "memory");
    }
    __syncthreads();

    // ---- fragments in contiguous K order (K = 16*kgrp + j), same for A and B
    v16h afr[2], bfr[4];
    #pragma unroll
    for (int mi = 0; mi < 2; ++mi) {
      const half_t* src = &As[buf][wm * 32 + mi * 16 + l16][0];
      const int kb = kgrp * 16;
      #pragma unroll
      for (int j = 0; j < 8; ++j)  afr[mi][j] = src[kb + j];        // K = kb+j
      #pragma unroll
      for (int j = 8; j < 16; ++j) afr[mi][j] = src[kb + j];        // K = kb+j
    }
    #pragma unroll
    for (int ni = 0; ni < 4; ++ni) {
      const half_t* src = &Bs[buf][wn * 64 + ni * 16 + l16][kgrp * 16];
      #pragma unroll
      for (int j = 0; j < 16; ++j) bfr[ni][j] = src[j];             // K = kgrp*16+j
    }

    // ---- 8 WMMAs: 32x64 wave tile advanced by K=32
    #pragma unroll
    for (int mi = 0; mi < 2; ++mi)
      #pragma unroll
      for (int ni = 0; ni < 4; ++ni)
        acc[mi][ni] = __builtin_amdgcn_wmma_f32_16x16x32_f16(
            false, afr[mi], false, bfr[ni],
            (short)0, acc[mi][ni], false, false);
    __syncthreads();   // all waves done reading buf before stage t+2 overwrites
  }

  // ---- epilogue: C layout VGPR r -> M = r + 8*kgrp, N = lane&15
  #pragma unroll
  for (int mi = 0; mi < 2; ++mi) {
    #pragma unroll
    for (int ni = 0; ni < 4; ++ni) {
      const int rbase = wm * 32 + mi * 16 + 8 * kgrp;
      const int col   = colBlock + wn * 64 + ni * 16 + l16;
      #pragma unroll
      for (int r = 0; r < 8; ++r) {
        long off = (rowBlock + rbase + r) * (long)ldc + col;
        float v = acc[mi][ni][r];
        if (mode == 0)      Cf[off] = v;
        else if (mode == 1) Ch[off] = (half_t)v;
        else                Out[off] = fmaxf(v + bias[col], 0.f);
      }
    }
  }
}

// ---------------------------------------------------------------- launcher
extern "C" void kernel_launch(void* const* d_in, const int* in_sizes, int n_in,
                              void* d_out, int out_size, void* d_ws, size_t ws_size,
                              hipStream_t stream) {
  const float* x        = (const float*)d_in[0];   // [8192,1024]
  const float* W        = (const float*)d_in[1];   // [1024,8192]
  const float* b        = (const float*)d_in[2];   // [1024]
  const int*   idx      = (const int*)d_in[3];     // [4096]
  const int*   idx_comp = (const int*)d_in[4];     // [4096]
  float*       out      = (float*)d_out;           // [8192,1024]

  // workspace layout (~321 MB)
  char* w = (char*)d_ws;
  float*  amax = (float*)w;                              // [0]=absmax(x) [1]=absmax(hexp)
  half_t* xq   = (half_t*)(w + 256);                     // 16 MB  quant(x) f16
  half_t* xf   = xq  + (size_t)Bb * Nn;                  // 16 MB  x as f16
  half_t* Wq   = xf  + (size_t)Bb * Nn;                  //  8 MB  fake-quant W_exp [Nn][Mh]
  half_t* WqT  = Wq  + (size_t)Nn * Mh;                  //  8 MB  Wq transposed [Mh][Nn]
  half_t* Wc   = WqT + (size_t)Nn * Mh;                  //  8 MB  W_cont [Nn][Mh]
  half_t* WcT  = Wc  + (size_t)Nn * Mh;                  //  8 MB  Wc transposed [Mh][Nn]
  half_t* hc   = WcT + (size_t)Nn * Mh;                  // 64 MB  hidden_cont f16
  half_t* hq   = hc  + (size_t)Bb * Mh;                  // 64 MB  quant(hidden_exp) f16
  float*  hexp = (float*)(hq + (size_t)Bb * Mh);         // 128 MB hidden_exp f32

  k_init_scalars<<<1, 1, 0, stream>>>(amax);

  // absmax(x), quantize x, cast x
  k_absmax<<<1024, 256, 0, stream>>>(x, (long)Bb * Nn, (unsigned*)&amax[0]);
  k_act_quant<<<2048, 256, 0, stream>>>(x, &amax[0], xq, xf, (long)Bb * Nn);

  // weight paths (both orientations)
  k_weight_quant<<<(Nn * Mh / 32) / 8, 256, 0, stream>>>(W, idx, Wq, WqT);
  k_weight_cont<<<2048, 256, 0, stream>>>(W, idx_comp, Wc, WcT);

  // layer 1: hidden_exp(f32) = xq @ Wq ; hidden_cont(f16) = xf @ Wc
  // B^T layout: WqT/WcT are [Mh][Nn], row length = K = Nn
  dim3 g1(Mh / 128, Bb / 128);
  k_gemm_wmma<<<g1, 256, 0, stream>>>(xq, WqT, nullptr, nullptr,
                                      Nn, hexp, nullptr, nullptr,
                                      nullptr, Mh, 0);
  k_gemm_wmma<<<g1, 256, 0, stream>>>(xf, WcT, nullptr, nullptr,
                                      Nn, nullptr, hc, nullptr,
                                      nullptr, Mh, 1);

  // absmax(hidden_exp) + quantize to f16
  k_absmax<<<2048, 256, 0, stream>>>(hexp, (long)Bb * Mh, (unsigned*)&amax[1]);
  k_act_quant<<<4096, 256, 0, stream>>>(hexp, &amax[1], hq, nullptr, (long)Bb * Mh);

  // layer 2 (fused dual-GEMM): out = relu(hq@Wq^T + hc@Wc^T + b)
  // B^T layout: Wq/Wc are [Nn][Mh], row length = K = Mh
  dim3 g2(Nn / 128, Bb / 128);
  k_gemm_wmma<<<g2, 256, 0, stream>>>(hq, Wq, hc, Wc,
                                      Mh, nullptr, nullptr, b,
                                      out, Nn, 2);
}
